// LSNN_35313221107905
// MI455X (gfx1250) — matrix operand order repaired and checked by reading, not compiled, not run on hardware
//
#include <hip/hip_runtime.h>
#include <hip/hip_bf16.h>

// ---------------------------------------------------------------------------
// LSNN forward on MI455X (gfx1250, wave32, WMMA).
//
// Shapes: B=512, T=1024, ORDER=128 -> K=2*ORDER=256, H=1024, C=10.
// Grid: 32 batch tiles (M=16) x 4 hidden partitions (256 cols) = 128 WGs,
// 256 threads (8 waves) each. Each wave owns two 16-wide column tiles of v_h
// kept in WMMA f32 accumulators for the whole T loop; W_hdn is pre-packed to
// f16 in the exact WMMA B-fragment lane layout and held resident in VGPRs.
// Output integrator is linear -> each H-partition runs its own beta-scan of
// partial v_o and atomically adds into d_out (zeroed first each launch).
//
// Round-2 changes (schedule only):
//  * A-fragment ping-pong: LDS load for k-block kb+1 issued before the two
//    WMMAs of kb, removing the s_wait_dscnt 0 before every WMMA pair.
//  * Software-pipelined x: loads for step t+1 issued during step t's compute;
//    __builtin_prefetch moved out to t+2 (HBM->L2 stage).
// ---------------------------------------------------------------------------

typedef __attribute__((ext_vector_type(16))) _Float16 v16h;
typedef __attribute__((ext_vector_type(8)))  _Float16 v8h;
typedef __attribute__((ext_vector_type(8)))  float    v8f;

#define B_     512
#define T_     1024
#define ORDER_ 128
#define KDIM_  256      // 2*ORDER
#define H_     1024
#define C_     10
#define THR_   1.0f
#define ALPHA_ 0.9f
#define BETA_  0.9f

// packed W_hdn: [ntile(64)][kb(8)][lane(32)][half(16)]  (f16)
#define WP_ELEMS  (64 * 8 * 32 * 16)      // 262144 halves
// packed W_otp: [kb(32)][lane(32)][half(16)]            (f16, cols 10..15 = 0)
#define WOP_ELEMS (32 * 32 * 16)          // 16384 halves

// ---------------------------------------------------------------------------
__global__ void lsnn_zero_out(float* __restrict__ out, int n) {
    for (int i = blockIdx.x * blockDim.x + threadIdx.x; i < n;
         i += gridDim.x * blockDim.x)
        out[i] = 0.0f;
}

// Pack W_hdn (f32 row-major [K=256][H=1024]) into f16 WMMA B fragments.
// B layout (16x16x32 f16, K x N = 32x16): lanes 0-15 hold col N=lane with
// halves h=0..15 -> K = kb*32 + h; lanes 16-31 hold N=lane-16 with
// K = kb*32 + 16 + h  (contiguous-K per half-wave; this code's order, same as the A loader).
__global__ void lsnn_pack_whdn(const float* __restrict__ Wh,
                               _Float16* __restrict__ Wp) {
    int e = blockIdx.x * blockDim.x + threadIdx.x;
    if (e >= WP_ELEMS) return;
    int h     = e & 15;
    int lane  = (e >> 4) & 31;
    int kb    = (e >> 9) & 7;
    int ntile = e >> 12;
    int n = ntile * 16 + (lane & 15);
    int k = kb * 32 + ((lane >> 4) ? 16 : 0) + h;
    Wp[e] = (_Float16)Wh[k * H_ + n];
}

// Pack W_otp (f32 [H=1024][C=10]) into f16 B fragments, N padded to 16.
__global__ void lsnn_pack_wotp(const float* __restrict__ Wo,
                               _Float16* __restrict__ Wop) {
    int e = blockIdx.x * blockDim.x + threadIdx.x;
    if (e >= WOP_ELEMS) return;
    int h    = e & 15;
    int lane = (e >> 4) & 31;
    int kb   = e >> 9;                 // 0..31 over H
    int n = lane & 15;
    int k = kb * 32 + ((lane >> 4) ? 16 : 0) + h;
    float v = (n < C_) ? Wo[k * C_ + n] : 0.0f;
    Wop[e] = (_Float16)v;
}

// ---------------------------------------------------------------------------
__global__ __launch_bounds__(256, 1)
void lsnn_main(const float* __restrict__ x,
               const _Float16* __restrict__ Wp,
               const _Float16* __restrict__ Wop,
               float* __restrict__ out) {
    // se: spike tile 16 rows x 256 K (f16), row stride 264 halves (528 B,
    // keeps 16B alignment and rotates LDS banks by 4 per row).
    __shared__ _Float16 se[16 * 264];
    // sh: per-wave 16x32 s_h staging (D-layout -> A-layout transpose),
    // row stride 40 halves (80 B, 16B aligned).
    __shared__ _Float16 sh[8 * 16 * 40];
    // voacc: 16x16 f32 cross-wave output reduction tile.
    __shared__ float voacc[256];

    const int tid     = threadIdx.x;
    const int w       = tid >> 5;          // wave 0..7
    const int lane    = tid & 31;
    const int lmod    = lane & 15;
    const int hihalf  = lane >> 4;         // 0 or 1
    const int b0      = blockIdx.x * 16;   // batch tile base
    const int hp      = blockIdx.y;        // hidden partition 0..3

    // ---- persistent W_hdn B fragments (2 n-tiles x 8 k-blocks) in VGPRs ----
    v16h wB[2][8];
#pragma unroll
    for (int nt = 0; nt < 2; ++nt) {
        const int ntile = hp * 16 + 2 * w + nt;
#pragma unroll
        for (int kb = 0; kb < 8; ++kb)
            wB[nt][kb] = *(const v16h*)(Wp + (((size_t)ntile * 8 + kb) * 32 + lane) * 16);
    }
    // W_otp fragment: wave w's 32 hidden cols == K-block (hp*8 + w).
    const v16h wO = *(const v16h*)(Wop + (((size_t)(hp * 8 + w)) * 32 + lane) * 16);

    // ---- persistent state ----
    v8f acc[2];                 // v_h for the two 16-col tiles
    v8f vo;                     // partial output integrator
#pragma unroll
    for (int i = 0; i < 8; ++i) { acc[0][i] = 0.f; acc[1][i] = 0.f; vo[i] = 0.f; }
    float ve[16];               // encoder membrane: thread owns K-column tid
#pragma unroll
    for (int r = 0; r < 16; ++r) ve[r] = 0.f;

    const int  kc   = tid;              // encoder K column 0..255
    const int  xcol = kc & (ORDER_ - 1);
    const bool neg  = (kc >= ORDER_);   // wave-uniform (32 lanes per half)

    // A-fragment loader: lane holds row lmod; K chunks at
    // kb*32 + hihalf*16 + [0..7] and + [8..15]  (contiguous-K, matches B).
    auto loadA = [&](int kb) -> v16h {
        const _Float16* ap = &se[lmod * 264 + kb * 32 + hihalf * 16];
        const v8h lo = *(const v8h*)ap;
        const v8h hi = *(const v8h*)(ap + 8);
        return __builtin_shufflevector(lo, hi,
            0, 1, 2, 3, 4, 5, 6, 7, 8, 9, 10, 11, 12, 13, 14, 15);
    };

    // ---- software-pipelined x: xcur holds step t, xnext loads step t+1 ----
    float xcur[16], xnext[16];
#pragma unroll
    for (int r = 0; r < 16; ++r) {
        xnext[r] = 0.f;
        xcur[r]  = x[(((size_t)(b0 + r) * T_) * ORDER_) + xcol];   // t = 0
    }

    for (int t = 0; t < T_; ++t) {
        voacc[tid] = 0.0f;              // zero reduction tile (barrier below)

        // HBM->L2 stage: prefetch step t+2 (global_prefetch_b8)
        if (t + 2 < T_) {
            const int pr = tid >> 4;          // row 0..15
            const int pc = (tid & 15) * 8;    // 32B strides across the row
            __builtin_prefetch(
                x + (((size_t)(b0 + pr) * T_ + (t + 2)) * ORDER_ + pc), 0, 1);
        }

        // ---- encoder: IF neuron with soft reset, spikes -> LDS (f16) ----
#pragma unroll
        for (int r = 0; r < 16; ++r) {
            const float cur = neg ? fmaxf(-xcur[r], 0.f) : fmaxf(xcur[r], 0.f);
            const float v   = ve[r] + cur;
            const float s   = (v >= THR_) ? 1.f : 0.f;
            ve[r] = v - s * THR_;
            se[r * 264 + kc] = (_Float16)s;
        }

        // L2->VGPR stage: issue loads for step t+1; a whole compute phase
        // (WMMAs + two barriers) covers their latency.
        if (t + 1 < T_) {
#pragma unroll
            for (int r = 0; r < 16; ++r)
                xnext[r] = x[(((size_t)(b0 + r) * T_ + (t + 1)) * ORDER_) + xcol];
        }
        __syncthreads();

        // ---- hidden layer: v_h = ALPHA*v_h + s_e @ W_hdn  (WMMA) ----
#pragma unroll
        for (int i = 0; i < 8; ++i) { acc[0][i] *= ALPHA_; acc[1][i] *= ALPHA_; }

        // ping-pong A fragments so the LDS load for kb+1 overlaps the two
        // WMMAs of kb (avoids s_wait_dscnt 0 before every WMMA pair).
        v16h a0 = loadA(0);
#pragma unroll
        for (int kb = 0; kb < 8; ++kb) {
            v16h a1;
            if (kb < 7) a1 = loadA(kb + 1);
            acc[0] = __builtin_amdgcn_wmma_f32_16x16x32_f16(
                false, a0, false, wB[0][kb], (short)0, acc[0], false, false);
            acc[1] = __builtin_amdgcn_wmma_f32_16x16x32_f16(
                false, a0, false, wB[1][kb], (short)0, acc[1], false, false);
            if (kb < 7) a0 = a1;
        }

        // ---- hidden spikes, soft reset; stage s_h (D-layout -> LDS) ----
        _Float16* shw = &sh[w * 16 * 40];
#pragma unroll
        for (int nt = 0; nt < 2; ++nt) {
#pragma unroll
            for (int i = 0; i < 8; ++i) {
                const float v = acc[nt][i];
                const float s = (v >= THR_) ? 1.f : 0.f;
                acc[nt][i] = v - s * THR_;
                shw[(i + hihalf * 8) * 40 + nt * 16 + lmod] = (_Float16)s;
            }
        }
        __syncthreads();

        // ---- output layer: vo = BETA*vo + s_h_block @ W_otp_block (WMMA) ----
#pragma unroll
        for (int i = 0; i < 8; ++i) vo[i] *= BETA_;
        {
            const _Float16* ap = &shw[lmod * 40 + hihalf * 16];
            const v8h lo = *(const v8h*)ap;
            const v8h hi = *(const v8h*)(ap + 8);
            const v16h a = __builtin_shufflevector(lo, hi,
                0, 1, 2, 3, 4, 5, 6, 7, 8, 9, 10, 11, 12, 13, 14, 15);
            vo = __builtin_amdgcn_wmma_f32_16x16x32_f16(
                false, a, false, wO, (short)0, vo, false, false);
        }
        // reduce the 8 wave partials in LDS (ds_add_f32)
#pragma unroll
        for (int i = 0; i < 8; ++i)
            atomicAdd(&voacc[(i + hihalf * 8) * 16 + lmod], vo[i]);
        __syncthreads();

        // one wave commits the 16x10 partial v_o(t) to global (atomic across
        // the 4 H-partitions; out is zeroed at launch start).
        if (tid < 32) {
#pragma unroll
            for (int i = 0; i < 5; ++i) {
                const int idx = tid + 32 * i;          // 0..159
                const int r = idx / C_, c = idx % C_;
                atomicAdd(out + (((size_t)(b0 + r) * T_ + t) * C_ + c),
                          voacc[r * 16 + c]);
            }
        }

        // rotate the x pipeline
#pragma unroll
        for (int r = 0; r < 16; ++r) xcur[r] = xnext[r];

        __syncthreads();   // protect voacc before next step's zeroing
    }
}

// ---------------------------------------------------------------------------
extern "C" void kernel_launch(void* const* d_in, const int* in_sizes, int n_in,
                              void* d_out, int out_size, void* d_ws, size_t ws_size,
                              hipStream_t stream) {
    const float* x  = (const float*)d_in[0];   // (512,1024,128) f32
    const float* Wh = (const float*)d_in[1];   // (256,1024) f32
    const float* Wo = (const float*)d_in[2];   // (1024,10) f32
    float* out = (float*)d_out;                // (512,1024,10) f32

    _Float16* Wp  = (_Float16*)d_ws;           // 512 KB packed W_hdn
    _Float16* Wop = Wp + WP_ELEMS;             // +32 KB packed W_otp

    lsnn_zero_out<<<2048, 256, 0, stream>>>(out, out_size);
    lsnn_pack_whdn<<<(WP_ELEMS + 255) / 256, 256, 0, stream>>>(Wh, Wp);
    lsnn_pack_wotp<<<(WOP_ELEMS + 255) / 256, 256, 0, stream>>>(Wo, Wop);

    dim3 grid(B_ / 16, 4);                     // 32 batch tiles x 4 H-partitions
    lsnn_main<<<grid, 256, 0, stream>>>(x, Wp, Wop, out);
}
